// GraphTransformerEncoder_35880156791231
// MI455X (gfx1250) — compile-verified, matrix operand order checked by reading, not run on hardware
//
#include <hip/hip_runtime.h>
#include <cstdint>

#define N_GRAPHS 128
#define NODES_PER_GRAPH 512
#define N_NODES (N_GRAPHS * NODES_PER_GRAPH)     // 65536
#define EDGES_PER_GRAPH 4096
#define N_EDGES (N_GRAPHS * EDGES_PER_GRAPH)     // 524288
#define D 128
#define H 8
#define DH 16
#define L_LAYERS 3
#define EPS 1e-5f

typedef __bf16 v16bf __attribute__((ext_vector_type(16)));
typedef __bf16 v8bf  __attribute__((ext_vector_type(8)));
typedef float  v8f   __attribute__((ext_vector_type(8)));

// ---------------------------------------------------------------------------
// f32 -> bf16 bulk convert (8 elements / thread, 16B stores)
// ---------------------------------------------------------------------------
__global__ __launch_bounds__(256) void cvt_bf16_kernel(
    const float* __restrict__ src, __bf16* __restrict__ dst, long long n8)
{
    long long i = (long long)blockIdx.x * blockDim.x + threadIdx.x;
    if (i >= n8) return;
    const float4 f0 = *(const float4*)(src + i * 8);
    const float4 f1 = *(const float4*)(src + i * 8 + 4);
    v8bf o;
    o[0] = (__bf16)f0.x; o[1] = (__bf16)f0.y; o[2] = (__bf16)f0.z; o[3] = (__bf16)f0.w;
    o[4] = (__bf16)f1.x; o[5] = (__bf16)f1.y; o[6] = (__bf16)f1.z; o[7] = (__bf16)f1.w;
    *(v8bf*)(dst + i * 8) = o;
}

// ---------------------------------------------------------------------------
// Weight convert + transpose: W[K=128][N=128] f32 -> Wt[N][K] bf16
// ---------------------------------------------------------------------------
__global__ __launch_bounds__(256) void cvt_wt_kernel(
    const float* __restrict__ W, __bf16* __restrict__ Wt)
{
    for (int i = blockIdx.x * 256 + threadIdx.x; i < D * D; i += gridDim.x * 256) {
        int k = i >> 7, n = i & 127;
        Wt[n * D + k] = (__bf16)W[k * D + n];
    }
}

// ---------------------------------------------------------------------------
// C[M x 128] = A[M x 128](bf16) @ W (Wt = N-major bf16 [128][128]) + bias
// Block: 256 threads = 8 waves; tile = 64 rows x 128 cols; K = 128.
// LDS-free. All 20 fragments (4 A + 16 B; 40 16-byte loads) issued up front so the
// compiler can use partial s_wait_loadcnt while earlier WMMAs execute.
// ---------------------------------------------------------------------------
__global__ __launch_bounds__(256) void gemm_wmma_kernel(
    const __bf16* __restrict__ A, const __bf16* __restrict__ Wt,
    const float* __restrict__ bias, float* __restrict__ C)
{
    const int tid     = threadIdx.x;
    const int wave    = tid >> 5;
    const int lane    = tid & 31;
    const int half    = lane >> 4;        // 0: lanes 0-15, 1: lanes 16-31
    const int rc      = lane & 15;        // row (A) / col (B)
    const int rowTile = (wave >> 1) * 16; // 0,16,32,48
    const int colBase = (wave & 1) * 64;  // 0 or 64

    const long long blockRow = (long long)blockIdx.x * 64;
    const __bf16* aRow = A + (blockRow + rowTile + rc) * D;

    // ---- issue every fragment load up front ----
    v16bf afrag[4];
    #pragma unroll
    for (int ks = 0; ks < 4; ++ks) {
        const int k0 = ks * 32 + 8 * half;
        v8bf alo = *(const v8bf*)(aRow + k0);
        v8bf ahi = *(const v8bf*)(aRow + k0 + 16);
        afrag[ks] = __builtin_shufflevector(alo, ahi,
            0, 1, 2, 3, 4, 5, 6, 7, 8, 9, 10, 11, 12, 13, 14, 15);
    }
    v16bf bfrag[4][4]; // [kstep][tile]
    #pragma unroll
    for (int ks = 0; ks < 4; ++ks) {
        const int k0 = ks * 32 + 8 * half;
        #pragma unroll
        for (int t = 0; t < 4; ++t) {
            const __bf16* bCol = Wt + (colBase + t * 16 + rc) * D;
            v8bf blo = *(const v8bf*)(bCol + k0);
            v8bf bhi = *(const v8bf*)(bCol + k0 + 16);
            bfrag[ks][t] = __builtin_shufflevector(blo, bhi,
                0, 1, 2, 3, 4, 5, 6, 7, 8, 9, 10, 11, 12, 13, 14, 15);
        }
    }

    // ---- 16 WMMAs: 4 independent accumulator chains ----
    v8f acc[4];
    #pragma unroll
    for (int t = 0; t < 4; ++t)
        #pragma unroll
        for (int j = 0; j < 8; ++j) acc[t][j] = 0.0f;

    #pragma unroll
    for (int ks = 0; ks < 4; ++ks)
        #pragma unroll
        for (int t = 0; t < 4; ++t)
            acc[t] = __builtin_amdgcn_wmma_f32_16x16x32_bf16(
                false, afrag[ks], false, bfrag[ks][t], (short)0, acc[t], false, false);

    // C/D layout: lane rc = col; VGPR r: row = r + 8*half
    #pragma unroll
    for (int t = 0; t < 4; ++t) {
        int col = colBase + t * 16 + rc;
        float b = bias ? bias[col] : 0.0f;
        #pragma unroll
        for (int r = 0; r < 8; ++r) {
            long long row = blockRow + rowTile + r + 8 * half;
            C[row * D + col] = acc[t][r] + b;
        }
    }
}

// ---------------------------------------------------------------------------
// Edge attention score: alpha[e,h] = 0.25 * dot(q[src], k[dst]+e[e]) (+ prev)
// ---------------------------------------------------------------------------
__global__ __launch_bounds__(256) void edge_alpha_kernel(
    const float* __restrict__ qn, const float* __restrict__ kn,
    const float* __restrict__ ep, const int* __restrict__ src,
    const int* __restrict__ dst, const float* __restrict__ aprev,
    float* __restrict__ alpha, int useprev)
{
    long long idx = (long long)blockIdx.x * blockDim.x + threadIdx.x;
    if (idx >= (long long)N_EDGES * H) return;
    int e = (int)(idx >> 3);
    int h = (int)(idx & 7);
    int s = src[e], d = dst[e];
    const float* qp  = qn + (long long)s * D + h * DH;
    const float* kp  = kn + (long long)d * D + h * DH;
    const float* epp = ep + (long long)e * D + h * DH;
    float acc = 0.0f;
    #pragma unroll
    for (int i = 0; i < DH; ++i) acc += qp[i] * (kp[i] + epp[i]);
    acc *= 0.25f; // 1/sqrt(DH), DH=16
    if (useprev) acc += aprev[idx];
    alpha[idx] = acc;
}

// order-preserving float<->uint key for atomic max
__device__ __forceinline__ unsigned f2key(float f) {
    unsigned u = __float_as_uint(f);
    return (u & 0x80000000u) ? ~u : (u | 0x80000000u);
}
__device__ __forceinline__ float key2f(unsigned k) {
    return __uint_as_float((k & 0x80000000u) ? (k & 0x7fffffffu) : ~k);
}

__global__ __launch_bounds__(256) void seg_max_kernel(
    const float* __restrict__ alpha, const int* __restrict__ dst,
    unsigned* __restrict__ mkey)
{
    long long idx = (long long)blockIdx.x * blockDim.x + threadIdx.x;
    if (idx >= (long long)N_EDGES * H) return;
    int e = (int)(idx >> 3);
    int h = (int)(idx & 7);
    atomicMax(&mkey[(long long)dst[e] * H + h], f2key(alpha[idx]));
}

__global__ __launch_bounds__(256) void seg_expsum_kernel(
    const float* __restrict__ alpha, const int* __restrict__ dst,
    const unsigned* __restrict__ mkey, float* __restrict__ aprob,
    float* __restrict__ ssum)
{
    long long idx = (long long)blockIdx.x * blockDim.x + threadIdx.x;
    if (idx >= (long long)N_EDGES * H) return;
    int e = (int)(idx >> 3);
    int h = (int)(idx & 7);
    int d = dst[e];
    float m  = key2f(mkey[(long long)d * H + h]);
    float av = __expf(alpha[idx] - m);
    aprob[idx] = av;
    atomicAdd(&ssum[(long long)d * H + h], av);
}

__global__ __launch_bounds__(256) void seg_norm_kernel(
    float* __restrict__ aprob, const int* __restrict__ dst,
    const float* __restrict__ ssum)
{
    long long idx = (long long)blockIdx.x * blockDim.x + threadIdx.x;
    if (idx >= (long long)N_EDGES * H) return;
    int e = (int)(idx >> 3);
    int h = (int)(idx & 7);
    aprob[idx] = aprob[idx] / ssum[(long long)dst[e] * H + h];
}

// ---------------------------------------------------------------------------
// Scatter messages: out[dst] += (v[src] + e) * a   (E*32 threads, float4 each)
// ---------------------------------------------------------------------------
__global__ __launch_bounds__(256) void message_kernel(
    const float* __restrict__ vn, const float* __restrict__ ep,
    const int* __restrict__ src, const int* __restrict__ dst,
    const float* __restrict__ aprob, float* __restrict__ out)
{
    long long idx = (long long)blockIdx.x * blockDim.x + threadIdx.x;
    if (idx >= (long long)N_EDGES * 32) return;
    int e = (int)(idx >> 5);
    int t = (int)(idx & 31);
    int s = src[e], d = dst[e];
    const float4 v  = *(const float4*)(vn + (long long)s * D + t * 4);
    const float4 ef = *(const float4*)(ep + (long long)e * D + t * 4);
    float a = aprob[(long long)e * H + (t >> 2)];
    float* o = out + (long long)d * D + t * 4;
    atomicAdd(o + 0, (v.x + ef.x) * a);
    atomicAdd(o + 1, (v.y + ef.y) * a);
    atomicAdd(o + 2, (v.z + ef.z) * a);
    atomicAdd(o + 3, (v.w + ef.w) * a);
}

// ---------------------------------------------------------------------------
// BatchNorm: column sums / sums of squares, then apply
// ---------------------------------------------------------------------------
__global__ __launch_bounds__(128) void bn_reduce_kernel(
    const float* __restrict__ out, float* __restrict__ sums,
    float* __restrict__ sumsq)
{
    int c = threadIdx.x;                        // 128 columns
    long long r0 = (long long)blockIdx.x * 256; // 256 blocks * 256 rows
    float s = 0.0f, sq = 0.0f;
    for (int r = 0; r < 256; ++r) {
        float v = out[(r0 + r) * D + c];
        s += v; sq += v * v;
    }
    atomicAdd(&sums[c], s);
    atomicAdd(&sumsq[c], sq);
}

__global__ __launch_bounds__(256) void bn_apply_kernel(
    const float* __restrict__ out, const float* __restrict__ sums,
    const float* __restrict__ sumsq, const float* __restrict__ gamma,
    const float* __restrict__ beta, float* __restrict__ xdst)
{
    long long idx = (long long)blockIdx.x * blockDim.x + threadIdx.x;
    if (idx >= (long long)N_NODES * D) return;
    int c = (int)(idx & 127);
    const float invN = 1.0f / (float)N_NODES;
    float mu  = sums[c] * invN;
    float var = sumsq[c] * invN - mu * mu;
    xdst[idx] = (out[idx] - mu) * rsqrtf(var + EPS) * gamma[c] + beta[c];
}

// ---------------------------------------------------------------------------
// Global mean pool: emb[g,c] = mean over 512 nodes
// ---------------------------------------------------------------------------
__global__ __launch_bounds__(128) void pool_kernel(
    const float* __restrict__ dense, float* __restrict__ emb)
{
    int g = blockIdx.x;   // 128 graphs
    int c = threadIdx.x;  // 128 features
    float s = 0.0f;
    for (int n = 0; n < NODES_PER_GRAPH; ++n)
        s += dense[((long long)g * NODES_PER_GRAPH + n) * D + c];
    emb[g * D + c] = s * (1.0f / (float)NODES_PER_GRAPH);
}

__global__ __launch_bounds__(256) void fill_u32_kernel(
    unsigned* __restrict__ p, unsigned v, long long n)
{
    long long i = (long long)blockIdx.x * blockDim.x + threadIdx.x;
    if (i < n) p[i] = v;
}

// ---------------------------------------------------------------------------
extern "C" void kernel_launch(void* const* d_in, const int* in_sizes, int n_in,
                              void* d_out, int out_size, void* d_ws, size_t ws_size,
                              hipStream_t stream) {
    (void)in_sizes; (void)n_in; (void)out_size; (void)ws_size;

    const float* x         = (const float*)d_in[0];
    const float* edge_feat = (const float*)d_in[1];
    const int*   eidx      = (const int*)d_in[2];
    // d_in[3] = batch (unused; equal-sized graphs)
    const float* Wk = (const float*)d_in[4];
    const float* bk = (const float*)d_in[5];
    const float* Wq = (const float*)d_in[6];
    const float* bq = (const float*)d_in[7];
    const float* Wv = (const float*)d_in[8];
    const float* bv = (const float*)d_in[9];
    const float* We = (const float*)d_in[10];
    const float* Ws = (const float*)d_in[11];
    const float* bs = (const float*)d_in[12];
    const float* gamma = (const float*)d_in[13];
    const float* beta  = (const float*)d_in[14];

    const int* src = eidx;
    const int* dst = eidx + N_EDGES;

    float* dense_out = (float*)d_out;                       // [N_NODES, D]
    float* emb_out   = dense_out + (long long)N_NODES * D;  // [N_GRAPHS, D]

    // ---- workspace carve ----
    const long long ND = (long long)N_NODES * D;   //  8.4M
    const long long ED = (long long)N_EDGES * D;   // 67.1M
    const long long EH = (long long)N_EDGES * H;   //  4.2M
    const long long NH = (long long)N_NODES * H;   //  0.5M
    float* p = (float*)d_ws;
    float* qb    = p; p += ND;
    float* kb_   = p; p += ND;
    float* vb    = p; p += ND;
    float* ob    = p; p += ND;     // skip + aggregated messages
    float* xb    = p; p += ND;     // intermediate x between layers
    float* ep    = p; p += ED;     // edge projections e = edge_feat @ We[l]
    float* alpha = p; p += EH;     // raw scores
    float* aprob = p; p += EH;     // normalized attention (== alpha_prev)
    unsigned* mkey = (unsigned*)p; p += NH;
    float* ssum  = p; p += NH;
    float* bns   = p; p += 128;    // bn sums (contiguous with bnsq)
    float* bnsq  = p; p += 128;
    // bf16 operand staging
    __bf16* xbf  = (__bf16*)p;  p += ND / 2;       // x in bf16     (16 MB)
    __bf16* efbf = (__bf16*)p;  p += ED / 2;       // edges in bf16 (128 MB)
    __bf16* wt   = (__bf16*)p;  p += (D * D) / 2;  // current weight, N-major bf16

    const int ehBlocks  = (int)(EH / 256);                       // 16384
    const int msgBlocks = (int)((long long)N_EDGES * 32 / 256);  // 65536

    // edge features -> bf16 once (constant across layers)
    cvt_bf16_kernel<<<(int)(ED / 8 / 256), 256, 0, stream>>>(edge_feat, efbf, ED / 8);

    for (int l = 0; l < L_LAYERS; ++l) {
        const float* xin = (l == 0) ? x : xb;
        const float* wkp = Wk + (long long)l * D * D;
        const float* wqp = Wq + (long long)l * D * D;
        const float* wvp = Wv + (long long)l * D * D;
        const float* wep = We + (long long)l * D * D;
        const float* wsp = Ws + (long long)l * D * D;

        // current x -> bf16 for WMMA A operand
        cvt_bf16_kernel<<<(int)(ND / 8 / 256), 256, 0, stream>>>(xin, xbf, ND / 8);

        // Node projections (per-node, hoisted out of per-edge gathers)
        cvt_wt_kernel<<<8, 256, 0, stream>>>(wkp, wt);
        gemm_wmma_kernel<<<N_NODES / 64, 256, 0, stream>>>(xbf, wt, bk + l * D, qb);
        cvt_wt_kernel<<<8, 256, 0, stream>>>(wqp, wt);
        gemm_wmma_kernel<<<N_NODES / 64, 256, 0, stream>>>(xbf, wt, bq + l * D, kb_);
        cvt_wt_kernel<<<8, 256, 0, stream>>>(wvp, wt);
        gemm_wmma_kernel<<<N_NODES / 64, 256, 0, stream>>>(xbf, wt, bv + l * D, vb);
        cvt_wt_kernel<<<8, 256, 0, stream>>>(wsp, wt);
        gemm_wmma_kernel<<<N_NODES / 64, 256, 0, stream>>>(xbf, wt, bs + l * D, ob);
        // Edge projection (the big GEMM: 524288 x 128 @ 128 x 128)
        cvt_wt_kernel<<<8, 256, 0, stream>>>(wep, wt);
        gemm_wmma_kernel<<<N_EDGES / 64, 256, 0, stream>>>(efbf, wt, nullptr, ep);

        // reset per-layer scratch
        fill_u32_kernel<<<(int)((NH + 255) / 256), 256, 0, stream>>>(mkey, 0u, NH);
        fill_u32_kernel<<<(int)((NH + 255) / 256), 256, 0, stream>>>((unsigned*)ssum, 0u, NH);
        fill_u32_kernel<<<1, 256, 0, stream>>>((unsigned*)bns, 0u, 256); // bns+bnsq

        // attention scores + segment softmax over incoming edges
        edge_alpha_kernel<<<ehBlocks, 256, 0, stream>>>(qb, kb_, ep, src, dst, aprob, alpha, l > 0);
        seg_max_kernel<<<ehBlocks, 256, 0, stream>>>(alpha, dst, mkey);
        seg_expsum_kernel<<<ehBlocks, 256, 0, stream>>>(alpha, dst, mkey, aprob, ssum);
        seg_norm_kernel<<<ehBlocks, 256, 0, stream>>>(aprob, dst, ssum);

        // aggregate messages onto skip-initialized out
        message_kernel<<<msgBlocks, 256, 0, stream>>>(vb, ep, src, dst, aprob, ob);

        // batch norm (training mode, biased variance)
        bn_reduce_kernel<<<256, 128, 0, stream>>>(ob, bns, bnsq);
        float* xdst = (l == L_LAYERS - 1) ? dense_out : xb;
        bn_apply_kernel<<<(int)(ND / 256), 256, 0, stream>>>(ob, bns, bnsq,
                                                             gamma + l * D, beta + l * D, xdst);
    }

    // graph-level mean pooling
    pool_kernel<<<N_GRAPHS, 128, 0, stream>>>(dense_out, emb_out);
}
